// Encoder_69818988363814
// MI455X (gfx1250) — matrix operand order repaired and checked by reading, not compiled, not run on hardware
//
#include <hip/hip_runtime.h>
#include <hip/hip_bf16.h>

typedef __bf16 bf16;
typedef __attribute__((ext_vector_type(16))) __bf16 bf16x16;
typedef __attribute__((ext_vector_type(8)))  float   f32x8;

union FragBF {
  bf16x16 v;
  uint4   q[2];
};

#define B_   64
#define T_   256
#define E_   512
#define U_   1024
#define U3_  3072
#define NBLK 64   // persistent blocks in GRU scan (must all be co-resident)

__device__ __forceinline__ bf16  f2b(float f) { return (bf16)f; }
__device__ __forceinline__ float b2f(bf16 h)  { return (float)h; }

// ---- CDNA5 async global->LDS copy (ASYNCcnt path, CDNA5 ISA) ----
typedef __attribute__((address_space(3))) void lds_void;

__device__ __forceinline__ void async_copy_b128(void* lds_dst, const void* gsrc) {
  lds_void* lp = (lds_void*)lds_dst;            // 32-bit LDS offset VGPR
  asm volatile("global_load_async_to_lds_b128 %0, %1, off"
               :
               : "v"(lp), "v"(gsrc)
               : "memory");
}

__device__ __forceinline__ void async_wait0() {
  asm volatile("s_wait_asynccnt 0" ::: "memory");
}

// -------------------- preprocessing --------------------

// x_bf16[b*T+t][e] = bf16(emb[tokens[b][t]][e])
__global__ void k_embed_bf16(const int* __restrict__ tokens,
                             const float* __restrict__ emb,
                             bf16* __restrict__ xb) {
  const int row = blockIdx.x;                 // b*T + t
  const int tok = tokens[row];
  const float* src = emb + (long)tok * E_;
  bf16* dst = xb + (long)row * E_;
  for (int e = threadIdx.x; e < E_; e += blockDim.x)
    dst[e] = f2b(src[e]);
}

// WT[n][k] = bf16(W[k][n]);  W is [K][N] row-major f32
__global__ void k_transpose_bf16(const float* __restrict__ W,
                                 bf16* __restrict__ WT,
                                 int K, int N) {
  long idx = (long)blockIdx.x * blockDim.x + threadIdx.x;
  if (idx >= (long)K * N) return;
  int n = (int)(idx % N);
  int k = (int)(idx / N);
  WT[(long)n * K + k] = f2b(W[idx]);
}

// h_bf16 init from f32 hidden; reset grid barrier
__global__ void k_init(const float* __restrict__ hidden,
                       bf16* __restrict__ hg,
                       unsigned* __restrict__ bar) {
  long idx = (long)blockIdx.x * blockDim.x + threadIdx.x;
  if (idx < (long)B_ * U_) hg[idx] = f2b(hidden[idx]);
  if (idx == 0) *bar = 0u;
}

// -------------------- phase 1: x_proj = x @ Wx + bx --------------------
// block = 256 threads = 8 waves; tile 128(M) x 64(N); K = 512
// B tile staged once per block into LDS via async DMA, stored k-chunk-major:
// sh_b[kc][n][8] so a 16-lane fragment load hits 16 consecutive 16B chunks.
__global__ __launch_bounds__(256)
void k_xproj(const bf16* __restrict__ x,     // [16384][512]
             const bf16* __restrict__ WxT,   // [3072][512]
             const float* __restrict__ bx,   // [3072]
             bf16* __restrict__ xp) {        // [16384][3072]
  __shared__ bf16 sh_b[E_ / 8][64][8];       // 64 KB

  const int lane = threadIdx.x & 31;
  const int wave = threadIdx.x >> 5;
  const int kh   = lane >> 4;                // K-half select
  const int lm   = lane & 15;

  const int m0 = blockIdx.y * 128 + wave * 16;
  const int n0 = blockIdx.x * 64;

  // async-stage B tile: 64 cols x 512 K bf16 = 4096 x 16B chunks
  for (int i = threadIdx.x; i < 64 * 64; i += 256) {
    const int kc = i & 63;                   // contiguous global reads per lane group
    const int n  = i >> 6;
    async_copy_b128(&sh_b[kc][n][0], WxT + (long)(n0 + n) * E_ + kc * 8);
  }

  const bf16* xa = x + (long)(m0 + lm) * E_;

  f32x8 acc[4];
  for (int c = 0; c < 4; ++c) acc[c] = (f32x8)0.0f;

  async_wait0();
  __syncthreads();

  for (int k = 0; k < E_; k += 32) {
    FragBF a;
    // A fragment in contiguous k order (k = 16*kh + s), same as B: chunks at kh*16 and kh*16+8 (8 contiguous bf16 = 16B)
    a.q[0] = *(const uint4*)(xa + k + kh * 16);
    a.q[1] = *(const uint4*)(xa + k + kh * 16 + 8);
    __builtin_prefetch(xa + k + 64, 0, 1);   // next A slab
    const int kc0 = (k >> 3) + kh * 2;       // B: lane-half selects K block of 16
#pragma unroll
    for (int c = 0; c < 4; ++c) {
      FragBF b;
      b.q[0] = *(const uint4*)&sh_b[kc0][c * 16 + lm][0];
      b.q[1] = *(const uint4*)&sh_b[kc0 + 1][c * 16 + lm][0];
      acc[c] = __builtin_amdgcn_wmma_f32_16x16x32_bf16(
          false, a.v, false, b.v, (short)0, acc[c], false, false);
    }
  }

  // C/D layout: VGPR r -> row r + 8*kh, col = lane&15
#pragma unroll
  for (int c = 0; c < 4; ++c) {
    const int col  = n0 + c * 16 + lm;
    const float bb = bx[col];
#pragma unroll
    for (int r = 0; r < 8; ++r) {
      const int row = m0 + r + kh * 8;
      xp[(long)row * U3_ + col] = f2b(acc[c][r] + bb);
    }
  }
}

// -------------------- phase 2: persistent GRU scan --------------------
// grid = (16 u-tiles, 4 batch-tiles) = 64 blocks; block = 384 threads = 12 waves
// wave j: gate g = j/4, 16-col chunk c = j%4 within the 64-wide u-tile
__global__ __launch_bounds__(384)
void k_gru(const bf16* __restrict__ xp,       // [16384][3072] bf16
           const bf16* __restrict__ WhT,      // [3072][1024] bf16 (row n = Wh col n)
           const float* __restrict__ bh,      // [3072]
           const float* __restrict__ hidden,  // [64][1024] f32
           bf16* __restrict__ hg,             // [64][1024] global h (exchanged via L2)
           unsigned* __restrict__ bar,
           float* __restrict__ out_seq,       // [64][256][1024]
           float* __restrict__ out_last) {    // [64][1024]
  __shared__ bf16  sh_h[16][1032];   // staged h tile (padded rows: 2064B stride)
  __shared__ float sh_rec[16][200];  // rec = h @ Wh for this block's 192 cols
  __shared__ float sh_hf[16][64];    // f32 running h for this (batch,u) tile

  const int tid  = threadIdx.x;
  const int lane = tid & 31;
  const int wave = tid >> 5;          // 0..11
  const int kh   = lane >> 4;
  const int lm   = lane & 15;

  const int ut = blockIdx.x;          // 0..15
  const int bt = blockIdx.y;          // 0..3
  const int m0 = bt * 16;             // batch row base
  const int u0 = ut * 64;             // u base

  const int g = wave >> 2;            // gate 0..2
  const int c = wave & 3;             // chunk 0..3
  const bf16* wb_base = WhT + (long)(g * U_ + u0 + c * 16 + lm) * U_ + kh * 16;

  for (int i = tid; i < 16 * 64; i += 384) {
    const int r = i >> 6, uu = i & 63;
    sh_hf[r][uu] = hidden[(long)(m0 + r) * U_ + u0 + uu];
  }

  for (int t = 0; t < T_; ++t) {
    // async-stage h[m0..m0+15][0..1023] bf16 into LDS (32KB, 2048 16B chunks)
    for (int cc = tid; cc < 2048; cc += 384) {
      const int r  = cc >> 7;
      const int k8 = (cc & 127) << 3;
      async_copy_b128(&sh_h[r][k8], &hg[(long)(m0 + r) * U_ + k8]);
    }
    async_wait0();
    __syncthreads();

    // rec tile: M=16 batch rows, N=16 cols (this wave), K=1024
    f32x8 acc = (f32x8)0.0f;
    const bf16* arow = &sh_h[lm][0];
    for (int k = 0; k < U_; k += 32) {
      FragBF a, b;
      a.q[0] = *(const uint4*)(arow + k + kh * 16);
      a.q[1] = *(const uint4*)(arow + k + kh * 16 + 8);
      b.q[0] = *(const uint4*)(wb_base + k);
      b.q[1] = *(const uint4*)(wb_base + k + 8);
      acc = __builtin_amdgcn_wmma_f32_16x16x32_bf16(
          false, a.v, false, b.v, (short)0, acc, false, false);
    }
    {
      const int coln = g * 64 + c * 16 + lm;
#pragma unroll
      for (int r = 0; r < 8; ++r)
        sh_rec[r + kh * 8][coln] = acc[r];
    }
    __syncthreads();

    // fused gates + state update for this (16 x 64) tile
    for (int i = tid; i < 16 * 64; i += 384) {
      const int r = i >> 6, uu = i & 63;
      const int b = m0 + r;
      const long xbase = ((long)b * T_ + t) * U3_;
      const float xz = b2f(xp[xbase + u0 + uu]);
      const float xr = b2f(xp[xbase + U_ + u0 + uu]);
      const float xh = b2f(xp[xbase + 2 * U_ + u0 + uu]);
      const float hz = sh_rec[r][uu]       + bh[u0 + uu];
      const float hr = sh_rec[r][64 + uu]  + bh[U_ + u0 + uu];
      const float hh = sh_rec[r][128 + uu] + bh[2 * U_ + u0 + uu];
      const float z    = 1.0f / (1.0f + __expf(-(xz + hz)));
      const float rr   = 1.0f / (1.0f + __expf(-(xr + hr)));
      const float cand = tanhf(xh + rr * hh);   // reset_after: r applied post-matmul
      const float hold = sh_hf[r][uu];
      const float hnew = z * hold + (1.0f - z) * cand;
      sh_hf[r][uu] = hnew;
      out_seq[((long)b * T_ + t) * U_ + u0 + uu] = hnew;
      hg[(long)b * U_ + u0 + uu] = f2b(hnew);
    }

    // device-wide barrier: release h stores, spin, acquire before next stage
    __threadfence();
    __syncthreads();
#if __has_builtin(__builtin_amdgcn_s_cluster_barrier)
    __builtin_amdgcn_s_cluster_barrier();  // NOP when not dispatched in clusters
#endif
    if (tid == 0) {
      const unsigned target = (unsigned)NBLK * (unsigned)(t + 1);
      __hip_atomic_fetch_add(bar, 1u, __ATOMIC_RELEASE, __HIP_MEMORY_SCOPE_AGENT);
      while (__hip_atomic_load(bar, __ATOMIC_ACQUIRE, __HIP_MEMORY_SCOPE_AGENT) < target)
        __builtin_amdgcn_s_sleep(2);
    }
    __syncthreads();
    __threadfence();
  }

  for (int i = tid; i < 16 * 64; i += 384) {
    const int r = i >> 6, uu = i & 63;
    out_last[(long)(m0 + r) * U_ + u0 + uu] = sh_hf[r][uu];
  }
}

// -------------------- launch --------------------

extern "C" void kernel_launch(void* const* d_in, const int* in_sizes, int n_in,
                              void* d_out, int out_size, void* d_ws, size_t ws_size,
                              hipStream_t stream) {
  const int*   tokens = (const int*)  d_in[0];
  const float* hidden = (const float*)d_in[1];
  const float* emb    = (const float*)d_in[2];
  const float* Wx     = (const float*)d_in[3];
  const float* bx     = (const float*)d_in[4];
  const float* Wh     = (const float*)d_in[5];
  const float* bh     = (const float*)d_in[6];
  (void)in_sizes; (void)n_in; (void)out_size; (void)ws_size;

  char* ws = (char*)d_ws;
  size_t o = 0;
  bf16* xb  = (bf16*)(ws + o); o += (size_t)B_ * T_ * E_ * 2;    // 16 MB
  bf16* WxT = (bf16*)(ws + o); o += (size_t)U3_ * E_ * 2;        //  3 MB
  bf16* WhT = (bf16*)(ws + o); o += (size_t)U3_ * U_ * 2;        //  6 MB
  bf16* xp  = (bf16*)(ws + o); o += (size_t)B_ * T_ * U3_ * 2;   // 100 MB
  bf16* hg  = (bf16*)(ws + o); o += (size_t)B_ * U_ * 2;         // 128 KB
  unsigned* bar = (unsigned*)(ws + o); o += 256;

  k_embed_bf16<<<B_ * T_, 256, 0, stream>>>(tokens, emb, xb);
  k_transpose_bf16<<<((E_ * U3_) + 255) / 256, 256, 0, stream>>>(Wx, WxT, E_, U3_);
  k_transpose_bf16<<<((U_ * U3_) + 255) / 256, 256, 0, stream>>>(Wh, WhT, U_, U3_);
  k_init<<<((B_ * U_) + 255) / 256, 256, 0, stream>>>(hidden, hg, bar);

  dim3 g1(U3_ / 64, (B_ * T_) / 128);     // (48, 128)
  k_xproj<<<g1, 256, 0, stream>>>(xb, WxT, bx, xp);

  float* out_seq  = (float*)d_out;
  float* out_last = out_seq + (size_t)B_ * T_ * U_;
  dim3 g2(U_ / 64, B_ / 16);              // (16, 4) = 64 persistent blocks
  k_gru<<<g2, 384, 0, stream>>>(xp, WhT, bh, hidden, hg, bar, out_seq, out_last);
}
